// OTCoupler_37374805409873
// MI455X (gfx1250) — matrix operand order repaired and checked by reading, not compiled, not run on hardware
//
#include <hip/hip_runtime.h>
#include <hip/hip_bf16.h>
#include <math.h>

// ---------------------------------------------------------------------------
// OT coupler: cost GEMM (WMMA f16-split -> f32), 100 log-Sinkhorn iters over a
// 256MB fp32 logK matrix, row argmax, gather.
//
// Roofline: Sinkhorn = 200 passes x 256MB. L2 = 192MB, so we pin rows
// [0, SPLIT_ROW) (~176MB) in L2 and stream the tail (~80MB) with TH=NT
// loads/stores -> per-pass HBM traffic ~80MB, total ~16GB ~ 0.7ms @ 23.3TB/s,
// vs 2.2ms fully-thrashing. All passes use b128 vector accesses.
// ---------------------------------------------------------------------------

typedef __attribute__((ext_vector_type(16))) _Float16 v16h;
typedef __attribute__((ext_vector_type(8)))  _Float16 v8h;
typedef __attribute__((ext_vector_type(8)))  float    v8f;
typedef __attribute__((ext_vector_type(4)))  float    v4f;

#define NROWS 8192
#define MCOLS 8192
#define DDIM  512
#define INV_EPS 10.0f      // 1/0.1
#define SINK_ITERS 100
#define COL_CHUNKS 32      // 8192 / 256 rows per partial block
#define SPLIT_ROW 5632     // rows < SPLIT_ROW kept L2-resident (176MB of 192MB)

// ---------------- cache-policy load/store helpers ---------------------------
template <bool NT>
__device__ __forceinline__ v4f ld4(const float* p) {
    const v4f* q = reinterpret_cast<const v4f*>(p);
    if constexpr (NT) return __builtin_nontemporal_load(q);
    else              return *q;
}

__device__ __forceinline__ void st1(float* p, float v, bool nt) {
    if (nt) __builtin_nontemporal_store(v, p);
    else    *p = v;
}

// ---------------- split fp32 -> (f16 hi, f16 lo residual) -------------------
__global__ void split_f32_f16(const float* __restrict__ x,
                              _Float16* __restrict__ hi,
                              _Float16* __restrict__ lo, int n) {
    int t = blockIdx.x * 256 + threadIdx.x;
    if (t < n) {
        float v = x[t];
        _Float16 h = (_Float16)v;
        hi[t] = h;
        lo[t] = (_Float16)(v - (float)h);
    }
}

// ---------------- per-row squared L2 norms (one wave32 per row) -------------
__global__ void row_norms(const float* __restrict__ x, float* __restrict__ out,
                          int rows) {
    int wid  = threadIdx.x >> 5;
    int lane = threadIdx.x & 31;
    int row  = blockIdx.x * 8 + wid;
    if (row >= rows) return;
    const float* p = x + (size_t)row * DDIM;
    float s = 0.0f;
    for (int k = lane; k < DDIM; k += 32) { float v = p[k]; s = fmaf(v, v, s); }
    #pragma unroll
    for (int off = 16; off; off >>= 1) s += __shfl_xor(s, off, 32);
    if (lane == 0) out[row] = s;
}

__global__ void zero2(float* __restrict__ a, float* __restrict__ b, int n) {
    int t = blockIdx.x * 256 + threadIdx.x;
    if (t < n) { a[t] = 0.0f; b[t] = 0.0f; }
}

// ---------------- A fragment loader: 16x32 A, contiguous K order (same as B) ------------------
// lane half lo: V0-7 = K 0..15
// lane half hi: V0-7 = K 16..31
__device__ __forceinline__ v16h load_afrag(const _Float16* __restrict__ rowp,
                                           int ko) {
    const v8h a = *reinterpret_cast<const v8h*>(rowp + ko);
    const v8h b = *reinterpret_cast<const v8h*>(rowp + ko + 8);
    v16h r;
    #pragma unroll
    for (int t = 0; t < 8; ++t) { r[t] = a[t]; r[t + 8] = b[t]; }
    return r;
}

// ---------------- GEMM: logK = (2*x0.x1^T - xsq_i - ysq_j) / eps ------------
// 256 threads = 8 waves; each wave owns one 16x16 tile; block tile 32x64.
__global__ void gemm_logK(const _Float16* __restrict__ x0h,
                          const _Float16* __restrict__ x0l,
                          const _Float16* __restrict__ x1h,
                          const _Float16* __restrict__ x1l,
                          const float* __restrict__ xsq,
                          const float* __restrict__ ysq,
                          float* __restrict__ logK) {
    const int w    = threadIdx.x >> 5;
    const int lane = threadIdx.x & 31;
    const int half = lane >> 4;
    const int lr   = lane & 15;
    const int ko   = half << 4;     // A K sub-offset: 0 or 16
    const int bko  = half << 4;     // B K sub-offset: 0 or 16

    const int rowBase = blockIdx.y * 32 + (w >> 2) * 16;
    const int colBase = blockIdx.x * 64 + (w & 3) * 16;

    const _Float16* arh = x0h + (size_t)(rowBase + lr) * DDIM;
    const _Float16* arl = x0l + (size_t)(rowBase + lr) * DDIM;
    const _Float16* brh = x1h + (size_t)(colBase + lr) * DDIM;
    const _Float16* brl = x1l + (size_t)(colBase + lr) * DDIM;

    v8f acc = {};
    for (int kc = 0; kc < DDIM; kc += 32) {
        v16h ah = load_afrag(arh + kc, ko);
        v16h al = load_afrag(arl + kc, ko);
        // B 32x16 f16: lane = column, 16 contiguous K per lane half
        v16h bh = *reinterpret_cast<const v16h*>(brh + kc + bko);
        v16h bl = *reinterpret_cast<const v16h*>(brl + kc + bko);
        // split-precision accumulate: hi*hi + hi*lo + lo*hi (~22-bit mantissa)
        acc = __builtin_amdgcn_wmma_f32_16x16x32_f16(false, al, false, bh,
                                                     (short)0, acc, false, false);
        acc = __builtin_amdgcn_wmma_f32_16x16x32_f16(false, ah, false, bl,
                                                     (short)0, acc, false, false);
        acc = __builtin_amdgcn_wmma_f32_16x16x32_f16(false, ah, false, bh,
                                                     (short)0, acc, false, false);
    }

    const int col  = colBase + lr;
    const float yq = ysq[col];
    // Tail rows are written non-temporally so rows [0, SPLIT_ROW) own L2.
    const bool nt = (rowBase >= SPLIT_ROW);
    #pragma unroll
    for (int v = 0; v < 8; ++v) {
        int row = rowBase + v + (half << 3);
        st1(&logK[(size_t)row * MCOLS + col],
            (2.0f * acc[v] - xsq[row] - yq) * INV_EPS, nt);
    }
}

// ---------------- online logsumexp accumulate over one row ------------------
template <bool NT>
__device__ __forceinline__ void row_accum(const float* __restrict__ row,
                                          const float* __restrict__ beta,
                                          float& m, float& s) {
    for (int j4 = threadIdx.x; j4 < MCOLS / 4; j4 += 256) {
        v4f v = ld4<NT>(row + j4 * 4);
        v4f b = *reinterpret_cast<const v4f*>(beta + j4 * 4);
        #pragma unroll
        for (int t = 0; t < 4; ++t) {
            float x = v[t] + b[t];
            if (x <= m) { s += __expf(x - m); }
            else        { s = s * __expf(m - x) + 1.0f; m = x; }
        }
    }
}

// ---------------- row pass: alpha[i] = -logsumexp_j(logK[i,j] + beta[j]) ----
__global__ void row_lse(const float* __restrict__ logK,
                        const float* __restrict__ beta,
                        float* __restrict__ alpha) {
    const int i = blockIdx.x;
    const float* row = logK + (size_t)i * MCOLS;
    float m = -__builtin_inff(), s = 0.0f;
    if (i < SPLIT_ROW) row_accum<false>(row, beta, m, s);
    else               row_accum<true >(row, beta, m, s);

    __shared__ float sm[256], ss[256];
    sm[threadIdx.x] = m; ss[threadIdx.x] = s;
    __syncthreads();
    for (int off = 128; off > 0; off >>= 1) {
        if (threadIdx.x < off) {
            float m2 = sm[threadIdx.x + off], s2 = ss[threadIdx.x + off];
            float m1 = sm[threadIdx.x],       s1 = ss[threadIdx.x];
            float mm = fmaxf(m1, m2);
            ss[threadIdx.x] = s1 * __expf(m1 - mm) + s2 * __expf(m2 - mm);
            sm[threadIdx.x] = mm;
        }
        __syncthreads();
    }
    if (threadIdx.x == 0) alpha[i] = -(sm[0] + __logf(ss[0]));
}

// ---------------- column pass stage 1: partial (max,sum), 4 cols/thread -----
template <bool NT>
__device__ __forceinline__ void col_accum(const float* __restrict__ p,
                                          const float* __restrict__ sa,
                                          v4f& m, v4f& s) {
    for (int r = 0; r < 256; ++r) {
        v4f v = ld4<NT>(p + (size_t)r * MCOLS);
        float a = sa[r];
        #pragma unroll
        for (int t = 0; t < 4; ++t) {
            float x = v[t] + a;
            if (x <= m[t]) { s[t] += __expf(x - m[t]); }
            else           { s[t] = s[t] * __expf(m[t] - x) + 1.0f; m[t] = x; }
        }
    }
}

__global__ void col_lse_partial(const float* __restrict__ logK,
                                const float* __restrict__ alpha,
                                float* __restrict__ pm, float* __restrict__ ps) {
    const int j0 = (blockIdx.x * 256 + threadIdx.x) * 4;
    const int r0 = blockIdx.y * 256;
    __shared__ float sa[256];
    sa[threadIdx.x] = alpha[r0 + threadIdx.x];
    __syncthreads();
    v4f m = { -__builtin_inff(), -__builtin_inff(),
              -__builtin_inff(), -__builtin_inff() };
    v4f s = {};
    const float* p = logK + (size_t)r0 * MCOLS + j0;
    if (r0 < SPLIT_ROW) col_accum<false>(p, sa, m, s);
    else                col_accum<true >(p, sa, m, s);
    *reinterpret_cast<v4f*>(pm + (size_t)blockIdx.y * MCOLS + j0) = m;
    *reinterpret_cast<v4f*>(ps + (size_t)blockIdx.y * MCOLS + j0) = s;
}

// ---------------- column pass stage 2: combine chunks -> beta ---------------
__global__ void col_lse_combine(const float* __restrict__ pm,
                                const float* __restrict__ ps,
                                float* __restrict__ beta) {
    const int j0 = (blockIdx.x * 256 + threadIdx.x) * 4;
    v4f m = { -__builtin_inff(), -__builtin_inff(),
              -__builtin_inff(), -__builtin_inff() };
    v4f s = {};
    for (int c = 0; c < COL_CHUNKS; ++c) {
        v4f m2 = *reinterpret_cast<const v4f*>(pm + (size_t)c * MCOLS + j0);
        v4f s2 = *reinterpret_cast<const v4f*>(ps + (size_t)c * MCOLS + j0);
        #pragma unroll
        for (int t = 0; t < 4; ++t) {
            float mm = fmaxf(m[t], m2[t]);
            s[t] = s[t] * __expf(m[t] - mm) + s2[t] * __expf(m2[t] - mm);
            m[t] = mm;
        }
    }
    v4f out;
    #pragma unroll
    for (int t = 0; t < 4; ++t) out[t] = -(m[t] + __logf(s[t]));
    *reinterpret_cast<v4f*>(beta + j0) = out;
}

// ---------------- argmax_j (logK[i,j] + beta[j]); first-max tie-break -------
template <bool NT>
__device__ __forceinline__ void amax_accum(const float* __restrict__ row,
                                           const float* __restrict__ beta,
                                           float& best, int& bj) {
    for (int j4 = threadIdx.x; j4 < MCOLS / 4; j4 += 256) {
        v4f v = ld4<NT>(row + j4 * 4);
        v4f b = *reinterpret_cast<const v4f*>(beta + j4 * 4);
        #pragma unroll
        for (int t = 0; t < 4; ++t) {
            float x = v[t] + b[t];
            if (x > best) { best = x; bj = j4 * 4 + t; }
        }
    }
}

__global__ void argmax_rows(const float* __restrict__ logK,
                            const float* __restrict__ beta,
                            int* __restrict__ idx) {
    const int i = blockIdx.x;
    const float* row = logK + (size_t)i * MCOLS;
    float best = -__builtin_inff(); int bj = 0;
    if (i < SPLIT_ROW) amax_accum<false>(row, beta, best, bj);
    else               amax_accum<true >(row, beta, best, bj);

    __shared__ float bm[256]; __shared__ int bi[256];
    bm[threadIdx.x] = best; bi[threadIdx.x] = bj;
    __syncthreads();
    for (int off = 128; off > 0; off >>= 1) {
        if (threadIdx.x < off) {
            float v2 = bm[threadIdx.x + off]; int j2 = bi[threadIdx.x + off];
            if (v2 > bm[threadIdx.x] ||
                (v2 == bm[threadIdx.x] && j2 < bi[threadIdx.x])) {
                bm[threadIdx.x] = v2; bi[threadIdx.x] = j2;
            }
        }
        __syncthreads();
    }
    if (threadIdx.x == 0) idx[i] = bi[0];
}

// ---------------- out[N*D + i*D + d] = x1[idx[i]*D + d] ---------------------
__global__ void gather_rows(const float* __restrict__ x1,
                            const int* __restrict__ idx,
                            float* __restrict__ out) {
    int t = blockIdx.x * 256 + threadIdx.x;   // 0 .. N*D
    int i = t >> 9;                           // / 512
    int d = t & 511;
    out[t] = x1[(size_t)idx[i] * DDIM + d];
}

// ---------------------------------------------------------------------------
extern "C" void kernel_launch(void* const* d_in, const int* in_sizes, int n_in,
                              void* d_out, int out_size, void* d_ws,
                              size_t ws_size, hipStream_t stream) {
    (void)in_sizes; (void)n_in; (void)out_size; (void)ws_size;
    const float* x0 = (const float*)d_in[0];
    const float* x1 = (const float*)d_in[1];
    float* out = (float*)d_out;

    char* ws = (char*)d_ws;
    size_t off = 0;
    float*    logK = (float*)(ws + off);    off += (size_t)NROWS * MCOLS * 4; // 256 MB
    _Float16* x0h  = (_Float16*)(ws + off); off += (size_t)NROWS * DDIM * 2;
    _Float16* x0l  = (_Float16*)(ws + off); off += (size_t)NROWS * DDIM * 2;
    _Float16* x1h  = (_Float16*)(ws + off); off += (size_t)MCOLS * DDIM * 2;
    _Float16* x1l  = (_Float16*)(ws + off); off += (size_t)MCOLS * DDIM * 2;
    float*    xsq  = (float*)(ws + off);    off += (size_t)NROWS * 4;
    float*    ysq  = (float*)(ws + off);    off += (size_t)MCOLS * 4;
    float*    alpha= (float*)(ws + off);    off += (size_t)NROWS * 4;
    float*    beta = (float*)(ws + off);    off += (size_t)MCOLS * 4;
    float*    pm   = (float*)(ws + off);    off += (size_t)COL_CHUNKS * MCOLS * 4;
    float*    ps   = (float*)(ws + off);    off += (size_t)COL_CHUNKS * MCOLS * 4;
    int*      idx  = (int*)(ws + off);      off += (size_t)NROWS * 4;

    // prep: f16 hi/lo splits, row norms, zero duals
    split_f32_f16<<<(NROWS * DDIM) / 256, 256, 0, stream>>>(x0, x0h, x0l, NROWS * DDIM);
    split_f32_f16<<<(MCOLS * DDIM) / 256, 256, 0, stream>>>(x1, x1h, x1l, MCOLS * DDIM);
    row_norms<<<NROWS / 8, 256, 0, stream>>>(x0, xsq, NROWS);
    row_norms<<<MCOLS / 8, 256, 0, stream>>>(x1, ysq, MCOLS);
    zero2<<<NROWS / 256, 256, 0, stream>>>(alpha, beta, NROWS);

    // logK once via WMMA (68.7 GFLOP -> negligible)
    gemm_logK<<<dim3(MCOLS / 64, NROWS / 32), 256, 0, stream>>>(
        x0h, x0l, x1h, x1l, xsq, ysq, logK);

    // 100 Sinkhorn iterations: 200 passes over logK; head rows L2-resident,
    // tail rows streamed with NT policy -> ~80MB HBM per pass.
    for (int it = 0; it < SINK_ITERS; ++it) {
        row_lse<<<NROWS, 256, 0, stream>>>(logK, beta, alpha);
        col_lse_partial<<<dim3(MCOLS / 1024, COL_CHUNKS), 256, 0, stream>>>(
            logK, alpha, pm, ps);
        col_lse_combine<<<MCOLS / 1024, 256, 0, stream>>>(pm, ps, beta);
    }

    // argmax + outputs
    argmax_rows<<<NROWS, 256, 0, stream>>>(logK, beta, idx);
    hipMemcpyAsync(out, x0, (size_t)NROWS * DDIM * sizeof(float),
                   hipMemcpyDeviceToDevice, stream);
    gather_rows<<<(NROWS * DDIM) / 256, 256, 0, stream>>>(
        x1, idx, out + (size_t)NROWS * DDIM);
}
